// AttentionWithVQ_65395172049438
// MI455X (gfx1250) — matrix operand order repaired and checked by reading, not compiled, not run on hardware
//
#include <hip/hip_runtime.h>
#include <hip/hip_bf16.h>

typedef _Float16 half_t;
typedef __attribute__((ext_vector_type(16))) _Float16 v16h;
typedef __attribute__((ext_vector_type(8)))  _Float16 v8h;
typedef __attribute__((ext_vector_type(8)))  float    v8f;

#define WMMA_F16(A, B, C) \
    __builtin_amdgcn_wmma_f32_16x16x32_f16(false, (A), false, (B), (short)0, (C), false, false)

// ---- problem constants: B=4, N=2048, DIM=512, HEAD=8, HEAD_DIM=64, VQE_K=3
static constexpr int  kN   = 2048;
static constexpr int  kDIM = 512;
static constexpr long kHSZ = 32L * 2048 * 64;        // elems of one [B,H,N,64] f16 plane
static constexpr long kXSZ = 8192L * 512;            // x (f16)
static constexpr long kWQ  = 1536L * 512;            // qkv_w (f16)
static constexpr long kWP  = 512L * 512;             // proj_w (f16)

__device__ inline v8f vzero8() {
    v8f z;
#pragma unroll
    for (int i = 0; i < 8; ++i) z[i] = 0.0f;
    return z;
}

// A-fragment (16x32 f16): lane m = lane&15, half = lane>>4.
// elems e <- K = kb + half*16 + e
__device__ inline v16h load_a_f16(const half_t* base, int row, int ld, int kb, int half8) {
    const half_t* p = base + (long)row * ld + kb + 2 * half8;
    v8h lo = *reinterpret_cast<const v8h*>(p);
    v8h hi = *reinterpret_cast<const v8h*>(p + 8);
    v16h r;
#pragma unroll
    for (int i = 0; i < 8; ++i) { r[i] = lo[i]; r[i + 8] = hi[i]; }
    return r;
}

// B-fragment (32x16 f16): lane n = lane&15; elems e <- K = kb + half*16 + e
__device__ inline v16h load_b_f16(const half_t* base, int row, int ld, int kb, int half16) {
    return *reinterpret_cast<const v16h*>(base + (long)row * ld + kb + half16);
}

// ---------------------------------------------------------------------------
// Kernel 0: pack fp32 operands (x, qkv_w, proj_w) to f16 once
// ---------------------------------------------------------------------------
__global__ __launch_bounds__(256) void pack_kernel(
    const float* __restrict__ x, const float* __restrict__ wqkv,
    const float* __restrict__ wproj, half_t* __restrict__ xh,
    half_t* __restrict__ wqkvh, half_t* __restrict__ wprojh) {
    long idx = (long)blockIdx.x * 256 + threadIdx.x;       // [0, 5242880)
    if (idx < kXSZ)                xh[idx]            = (half_t)x[idx];
    else if (idx < kXSZ + kWQ)     wqkvh[idx - kXSZ]  = (half_t)wqkv[idx - kXSZ];
    else                           wprojh[idx - kXSZ - kWQ] = (half_t)wproj[idx - kXSZ - kWQ];
}

// ---------------------------------------------------------------------------
// Kernel 1: depthwise 3-tap conv along N for m,s -> pe1h (x scale*0.5), pe2h
// ---------------------------------------------------------------------------
__global__ __launch_bounds__(256) void dwconv_kernel(
    const float* __restrict__ m, const float* __restrict__ s,
    const float* __restrict__ w1, const float* __restrict__ b1,
    const float* __restrict__ w2, const float* __restrict__ b2,
    half_t* __restrict__ pe1h, half_t* __restrict__ pe2h) {
    int idx = blockIdx.x * 256 + threadIdx.x;              // [0, 4*2048*512)
    int c  = idx & 511;
    int n  = (idx >> 9) & 2047;
    int b  = idx >> 20;
    int h  = c >> 6;
    int dh = c & 63;

    long base = ((long)(b * kN) + n) * kDIM + c;
    float mm1 = (n > 0)      ? m[base - kDIM] : 0.0f;
    float m0  = m[base];
    float mp1 = (n < kN - 1) ? m[base + kDIM] : 0.0f;
    float sm1 = (n > 0)      ? s[base - kDIM] : 0.0f;
    float s0  = s[base];
    float sp1 = (n < kN - 1) ? s[base + kDIM] : 0.0f;

    float y1 = (w1[h * 3 + 0] * mm1 + w1[h * 3 + 1] * m0 + w1[h * 3 + 2] * mp1 + b1[h]) * 0.0625f;
    float y2 =  w2[h * 3 + 0] * sm1 + w2[h * 3 + 1] * s0 + w2[h * 3 + 2] * sp1 + b2[h];

    long o = ((long)((b * 8 + h) * kN) + n) * 64 + dh;
    pe1h[o] = (half_t)y1;
    pe2h[o] = (half_t)y2;
}

// ---------------------------------------------------------------------------
// Kernel 2: qkv GEMM (8192x1536) = xh @ wqkvh^T ; 16x64 tile per wave.
// Writes qs (*scale*0.5), kh row-major per head; v transposed [B,H,64,N].
// ---------------------------------------------------------------------------
__global__ __launch_bounds__(256) void qkv_kernel(
    const half_t* __restrict__ xh, const half_t* __restrict__ wh,
    const float* __restrict__ bias,
    half_t* __restrict__ qs, half_t* __restrict__ kh, half_t* __restrict__ vT) {
    const int lane   = threadIdx.x & 31;
    const int wid    = blockIdx.x * 8 + (threadIdx.x >> 5); // [0, 512*24)
    const int im     = wid / 24, in = wid % 24;
    const int rowB   = im * 16, colG = in * 64;
    const int n15    = lane & 15;
    const int half8  = (lane >> 4) * 8;
    const int half16 = half8 * 2;

    v8f c[4];
#pragma unroll
    for (int t = 0; t < 4; ++t) c[t] = vzero8();

#pragma unroll 2
    for (int kk = 0; kk < 512; kk += 32) {
        v16h a = load_a_f16(xh, rowB + n15, 512, kk, half8);
#pragma unroll
        for (int t = 0; t < 4; ++t) {
            v16h b = load_b_f16(wh, colG + t * 16 + n15, 512, kk, half16);
            c[t] = WMMA_F16(a, b, c[t]);
        }
    }

    const int sec = colG >> 9;           // 0=q 1=k 2=v (uniform per wave)
    const int cc  = colG & 511;
    const int h   = cc >> 6;             // colG multiple of 64 -> one head
#pragma unroll
    for (int t = 0; t < 4; ++t) {
        const int dh = t * 16 + n15;
        const float bv = bias[colG + t * 16 + n15];
        if (sec == 2) {                  // v: rows contiguous in transposed layout
            v8h pk;
#pragma unroll
            for (int r = 0; r < 8; ++r) pk[r] = (half_t)(c[t][r] + bv);
            int row0 = rowB + half8;
            int b_   = row0 >> 11, nn = row0 & 2047;
            int hb   = (b_ << 3) | h;
            *reinterpret_cast<v8h*>(vT + ((long)hb * 64 + dh) * kN + nn) = pk;
        } else {
#pragma unroll
            for (int r = 0; r < 8; ++r) {
                int row = rowB + half8 + r;
                int b_  = row >> 11, nn = row & 2047;
                int hb  = (b_ << 3) | h;
                float v = c[t][r] + bv;
                if (sec == 0) qs[((long)hb * kN + nn) * 64 + dh] = (half_t)(v * 0.0625f);
                else          kh[((long)hb * kN + nn) * 64 + dh] = (half_t)v;
            }
        }
    }
}

// ---------------------------------------------------------------------------
// Kernel 3: flash attention. One wave = (b,h, 32 queries = 2 q-tiles);
// streams 32-key blocks. S^T = K·Q^T + PE2·PE1^T (keys = M so each lane owns
// one query column), online softmax, O^T = V^T·P^T. A-fragments (K/PE2/V)
// are reused across both query tiles -> halved L2 traffic.
// ---------------------------------------------------------------------------
__global__ __launch_bounds__(256) void attn_kernel(
    const half_t* __restrict__ qs, const half_t* __restrict__ khp,
    const half_t* __restrict__ vT, const half_t* __restrict__ p1h,
    const half_t* __restrict__ p2h, half_t* __restrict__ aout) {
    const int lane   = threadIdx.x & 31;
    const int wid    = blockIdx.x * 8 + (threadIdx.x >> 5); // [0, 2048)
    const int qpair  = wid & 63;                            // 64 pairs of 32 queries
    const int bh     = wid >> 6;
    const int b      = bh >> 3, h = bh & 7;
    const int n15    = lane & 15;
    const int half8  = (lane >> 4) * 8;
    const int half16 = half8 * 2;
    const bool h0    = (half8 == 0);
    const long hbase = (long)bh * kN * 64;

    const half_t* Q  = qs  + hbase;
    const half_t* K  = khp + hbase;
    const half_t* V  = vT  + hbase;
    const half_t* P1 = p1h + hbase;
    const half_t* P2 = p2h + hbase;

    // query-side B-fragments, held for the whole key loop: [qtile][dh-chunk]
    v16h bq[2][2], bp[2][2];
#pragma unroll
    for (int qt = 0; qt < 2; ++qt) {
        int qrow = qpair * 32 + qt * 16 + n15;
#pragma unroll
        for (int ch = 0; ch < 2; ++ch) {
            bq[qt][ch] = load_b_f16(Q,  qrow, 64, ch * 32, half16);
            bp[qt][ch] = load_b_f16(P1, qrow, 64, ch * 32, half16);
        }
    }

    v8f o[2][4];
#pragma unroll
    for (int qt = 0; qt < 2; ++qt)
#pragma unroll
        for (int t = 0; t < 4; ++t) o[qt][t] = vzero8();
    float m_i[2] = {-3.0e38f, -3.0e38f};
    float l_i[2] = {0.0f, 0.0f};

    for (int kb = 0; kb < kN; kb += 32) {
        // S^T tiles: s[qt][t] = keys kb+t*16..+15  x  queries of tile qt
        v8f s[2][2];
#pragma unroll
        for (int qt = 0; qt < 2; ++qt) { s[qt][0] = vzero8(); s[qt][1] = vzero8(); }
#pragma unroll
        for (int t = 0; t < 2; ++t) {
#pragma unroll
            for (int ch = 0; ch < 2; ++ch) {
                v16h ak = load_a_f16(K,  kb + t * 16 + n15, 64, ch * 32, half8);
                s[0][t] = WMMA_F16(ak, bq[0][ch], s[0][t]);
                s[1][t] = WMMA_F16(ak, bq[1][ch], s[1][t]);
                v16h ap = load_a_f16(P2, kb + t * 16 + n15, 64, ch * 32, half8);
                s[0][t] = WMMA_F16(ap, bp[0][ch], s[0][t]);
                s[1][t] = WMMA_F16(ap, bp[1][ch], s[1][t]);
            }
        }

        v16h bP[2];
#pragma unroll
        for (int qt = 0; qt < 2; ++qt) {
            // ---- online softmax (each lane owns one query column) ----
            float tmax = -3.0e38f;
#pragma unroll
            for (int r = 0; r < 8; ++r) tmax = fmaxf(tmax, fmaxf(s[qt][0][r], s[qt][1][r]));
            tmax = fmaxf(tmax, __shfl_xor(tmax, 16, 32));
            float mnew = fmaxf(m_i[qt], tmax);
            float corr = __expf(m_i[qt] - mnew);

            float p0a[8], p1a[8], rsum = 0.0f;
#pragma unroll
            for (int r = 0; r < 8; ++r) {
                p0a[r] = __expf(s[qt][0][r] - mnew);
                p1a[r] = __expf(s[qt][1][r] - mnew);
                rsum += p0a[r] + p1a[r];
            }
            l_i[qt] = l_i[qt] * corr + rsum;
            m_i[qt] = mnew;
#pragma unroll
            for (int t = 0; t < 4; ++t)
#pragma unroll
                for (int r = 0; r < 8; ++r) o[qt][t][r] *= corr;

            // repack P^T (C-layout) -> B-fragment (K = 32 keys) via lane-pair xchg
#pragma unroll
            for (int j = 0; j < 8; ++j) {
                float x0  = p0a[j], x1 = p1a[j];
                float x0x = __shfl_xor(x0, 16, 32);
                float x1x = __shfl_xor(x1, 16, 32);
                bP[qt][j]     = (half_t)(h0 ? x0  : x1x);
                bP[qt][j + 8] = (half_t)(h0 ? x0x : x1);
            }
        }

        // ---- O^T += V^T · P^T : V fragment shared across both query tiles ----
#pragma unroll
        for (int dt = 0; dt < 4; ++dt) {
            v16h av = load_a_f16(V, dt * 16 + n15, kN, kb, half8);
            o[0][dt] = WMMA_F16(av, bP[0], o[0][dt]);
            o[1][dt] = WMMA_F16(av, bP[1], o[1][dt]);
        }
    }

#pragma unroll
    for (int qt = 0; qt < 2; ++qt) {
        float lt  = l_i[qt] + __shfl_xor(l_i[qt], 16, 32);
        float inv = 1.0f / lt;
        int qrow = qpair * 32 + qt * 16 + n15;
        half_t* outp = aout + ((long)(b * kN + qrow)) * 512 + h * 64;
#pragma unroll
        for (int dt = 0; dt < 4; ++dt) {
            v8h pk;
#pragma unroll
            for (int r = 0; r < 8; ++r) pk[r] = (half_t)(o[qt][dt][r] * inv);
            *reinterpret_cast<v8h*>(outp + dt * 16 + half8) = pk;
        }
    }
}

// ---------------------------------------------------------------------------
// Kernel 4: output projection (8192x512) = aout(f16) @ wprojh^T + proj_b
// 16x64 tile per wave.
// ---------------------------------------------------------------------------
__global__ __launch_bounds__(256) void proj_kernel(
    const half_t* __restrict__ ain, const half_t* __restrict__ wh,
    const float* __restrict__ bias, float* __restrict__ out) {
    const int lane   = threadIdx.x & 31;
    const int wid    = blockIdx.x * 8 + (threadIdx.x >> 5); // [0, 512*8)
    const int im     = wid >> 3, in = wid & 7;
    const int rowB   = im * 16, colG = in * 64;
    const int n15    = lane & 15;
    const int half8  = (lane >> 4) * 8;
    const int half16 = half8 * 2;

    v8f c[4];
#pragma unroll
    for (int t = 0; t < 4; ++t) c[t] = vzero8();

#pragma unroll 2
    for (int kk = 0; kk < 512; kk += 32) {
        v16h a = load_a_f16(ain, rowB + n15, 512, kk, half8);
#pragma unroll
        for (int t = 0; t < 4; ++t) {
            v16h b = load_b_f16(wh, colG + t * 16 + n15, 512, kk, half16);
            c[t] = WMMA_F16(a, b, c[t]);
        }
    }
#pragma unroll
    for (int t = 0; t < 4; ++t) {
        int col = colG + t * 16 + n15;
        float bv = bias[col];
#pragma unroll
        for (int r = 0; r < 8; ++r)
            out[(long)(rowB + half8 + r) * 512 + col] = c[t][r] + bv;
    }
}

// ---------------------------------------------------------------------------
extern "C" void kernel_launch(void* const* d_in, const int* in_sizes, int n_in,
                              void* d_out, int out_size, void* d_ws, size_t ws_size,
                              hipStream_t stream) {
    const float* x      = (const float*)d_in[0];
    const float* m      = (const float*)d_in[1];
    const float* s      = (const float*)d_in[2];
    const float* qkv_w  = (const float*)d_in[3];
    const float* qkv_b  = (const float*)d_in[4];
    const float* proj_w = (const float*)d_in[5];
    const float* proj_b = (const float*)d_in[6];
    const float* pe1_w  = (const float*)d_in[7];
    const float* pe1_b  = (const float*)d_in[8];
    const float* pe2_w  = (const float*)d_in[9];
    const float* pe2_b  = (const float*)d_in[10];
    float* out = (float*)d_out;

    // workspace layout (f16): 6 planes of 8MB + packed x/weights (~10MB) = ~58MB
    half_t* qs     = (half_t*)d_ws;
    half_t* kh     = qs     + kHSZ;
    half_t* vT     = kh     + kHSZ;
    half_t* pe1h   = vT     + kHSZ;
    half_t* pe2h   = pe1h   + kHSZ;
    half_t* aout   = pe2h   + kHSZ;
    half_t* xh     = aout   + kHSZ;
    half_t* wqkvh  = xh     + kXSZ;
    half_t* wprojh = wqkvh  + kWQ;

    pack_kernel<<<20480, 256, 0, stream>>>(x, qkv_w, proj_w, xh, wqkvh, wprojh);
    dwconv_kernel<<<16384, 256, 0, stream>>>(m, s, pe1_w, pe1_b, pe2_w, pe2_b, pe1h, pe2h);
    qkv_kernel<<<1536, 256, 0, stream>>>(xh, wqkvh, qkv_b, qs, kh, vT);
    attn_kernel<<<256, 256, 0, stream>>>(qs, kh, vT, pe1h, pe2h, aout);
    proj_kernel<<<512, 256, 0, stream>>>(aout, wprojh, proj_b, out);

    (void)in_sizes; (void)n_in; (void)out_size; (void)ws_size;
}
